// HGT_72069551227212
// MI455X (gfx1250) — compile-verified, matrix operand order checked by reading, not run on hardware
//
#include <hip/hip_runtime.h>

// ---------------------------------------------------------------------------
// HGT layer for MI455X (gfx1250, wave32, WMMA).
// Matmul-shaped stages use v_wmma_f32_16x16x32_bf16 (codegen-confirmed).
// Per-type / per-relation weights are handled by computing all 4 variants and
// selecting per row (keeps the GEMM structure WMMA needs; compute is ~20 GF,
// far below the memory roofline: gathers hit the 192MB L2, HBM @ 23.3 TB/s).
// ---------------------------------------------------------------------------

typedef __bf16 bf16;
typedef __attribute__((ext_vector_type(16))) __bf16 v16bf;
typedef __attribute__((ext_vector_type(8)))  float  v8f;

#define DIM 128
#define NTY 4   // node types
#define NRE 4   // relation types
#define NH  8   // heads
#define DK  16  // head dim

static __device__ inline v8f wmma_bf16(v16bf a, v16bf b, v8f c) {
  // (neg_a, A, neg_b, B, c_mod, C, reuse_a, reuse_b)
  return __builtin_amdgcn_wmma_f32_16x16x32_bf16(false, a, false, b, (short)0, c,
                                                 false, false);
}

static __device__ inline v8f v8zero() {
  v8f v;
#pragma unroll
  for (int i = 0; i < 8; ++i) v[i] = 0.0f;
  return v;
}

// Monotonic float <-> uint mapping so atomicMax(uint) == float max.
static __device__ inline unsigned f2ord(float f) {
  unsigned u = __float_as_uint(f);
  return (u & 0x80000000u) ? ~u : (u | 0x80000000u);
}
static __device__ inline float ord2f(unsigned u) {
  unsigned v = (u & 0x80000000u) ? (u & 0x7fffffffu) : ~u;
  return __uint_as_float(v);
}

// ---------------------------------------------------------------- prep ------
__global__ void hgt_cvt_bf16(const float* __restrict__ src, bf16* __restrict__ dst,
                             long n) {
  long i = (long)blockIdx.x * blockDim.x + threadIdx.x;
  if (i < n) dst[i] = (bf16)src[i];
}

// dst[b][c][r] = src[b][r][c]   (B operand wants K contiguous per lane)
__global__ void hgt_transpose_bf16(const float* __restrict__ src,
                                   bf16* __restrict__ dst,
                                   int batch, int rows, int cols) {
  int i = blockIdx.x * blockDim.x + threadIdx.x;
  int total = batch * rows * cols;
  if (i >= total) return;
  int c = i % cols;
  int t = i / cols;
  int r = t % rows;
  int b = t / rows;
  dst[((long)b * cols + c) * rows + r] = (bf16)src[i];
}

// ------------------------------------------------- k,q,v projections --------
// Block = 256 (8 waves). blockIdx.x -> 16-node row tile, wave -> 16-col tile.
__global__ __launch_bounds__(256)
void hgt_project_kqv(const bf16* __restrict__ xbf,
                     const bf16* __restrict__ wT,   // [3][NTY][DIM(n)][DIM(k)]
                     const float* __restrict__ bk, const float* __restrict__ bq,
                     const float* __restrict__ bv,
                     const int* __restrict__ node_type,
                     float* __restrict__ kn, float* __restrict__ qn,
                     float* __restrict__ vn, int N) {
  const int wave = threadIdx.x >> 5;
  const int lane = threadIdx.x & 31;
  const int half = lane >> 4;
  const int lm   = lane & 15;
  const int m0 = blockIdx.x * 16;
  const int n0 = wave * 16;

  int arow = m0 + lm;
  if (arow >= N) arow = N - 1;
  v16bf a[4];
#pragma unroll
  for (int kc = 0; kc < 4; ++kc) {
    const bf16* px = xbf + (long)arow * DIM + kc * 32 + half * 16;
#pragma unroll
    for (int j = 0; j < 16; ++j) a[kc][j] = px[j];
  }

  int rowt[8];
#pragma unroll
  for (int j = 0; j < 8; ++j) {
    int r = m0 + half * 8 + j;
    rowt[j] = (r < N) ? node_type[r] : -1;
  }

  const float* biases[3] = {bk, bq, bv};
  float* outs[3] = {kn, qn, vn};

#pragma unroll
  for (int mat = 0; mat < 3; ++mat) {
    v8f res = v8zero();
#pragma unroll
    for (int t = 0; t < NTY; ++t) {
      v8f acc = v8zero();
#pragma unroll
      for (int kc = 0; kc < 4; ++kc) {
        const bf16* pw =
            wT + ((long)((mat * NTY + t) * DIM + (n0 + lm))) * DIM + kc * 32 + half * 16;
        v16bf b;
#pragma unroll
        for (int j = 0; j < 16; ++j) b[j] = pw[j];
        acc = wmma_bf16(a[kc], b, acc);
      }
      float bias = biases[mat][t * DIM + n0 + lm];
#pragma unroll
      for (int j = 0; j < 8; ++j)
        res[j] = (rowt[j] == t) ? (acc[j] + bias) : res[j];
    }
#pragma unroll
    for (int j = 0; j < 8; ++j) {
      int r = m0 + half * 8 + j;
      if (r < N) outs[mat][(long)r * DIM + n0 + lm] = res[j];
    }
  }
}

// ----------------------------------------------- attention scores -----------
// Block = 256: wave = head, blockIdx.x = 16-edge tile. K padded 16->32 (zeros).
__global__ __launch_bounds__(256)
void hgt_attn_scores(const float* __restrict__ kn, const float* __restrict__ qn,
                     const bf16* __restrict__ relAttT,   // [R][H][DK(n)][DK(d)]
                     const float* __restrict__ rel_pri,  // [R][H]
                     const int* __restrict__ edge_index, // [2][E]
                     const int* __restrict__ edge_type,
                     float* __restrict__ res_att,        // [E][H]
                     unsigned* __restrict__ seg_max_u,   // [N][H] ordered-uint
                     int E) {
  const int h    = threadIdx.x >> 5;
  const int lane = threadIdx.x & 31;
  const int half = lane >> 4;
  const int lm   = lane & 15;
  const long e0  = (long)blockIdx.x * 16;

  v16bf a;
  {
    long e = e0 + lm;
    if (e >= E) e = E - 1;
    const float* pk = kn + (long)edge_index[e] * DIM + h * DK;
#pragma unroll
    for (int j = 0; j < 16; ++j) a[j] = half ? (bf16)0.0f : (bf16)pk[j];
  }

  int et[8], dstc[8];
#pragma unroll
  for (int j = 0; j < 8; ++j) {
    long e = e0 + half * 8 + j;
    if (e >= E) e = E - 1;
    et[j]   = edge_type[e];
    dstc[j] = edge_index[(long)E + e];
  }

  v8f s = v8zero();
#pragma unroll
  for (int r = 0; r < NRE; ++r) {
    const bf16* pb = relAttT + ((long)(r * NH + h) * DK + lm) * DK;
    v16bf b;
#pragma unroll
    for (int j = 0; j < 16; ++j) b[j] = half ? (bf16)0.0f : pb[j];
    v8f acc = wmma_bf16(a, b, v8zero());
#pragma unroll
    for (int j = 0; j < 8; ++j) s[j] = (et[j] == r) ? acc[j] : s[j];
  }

  // row-dot with q (component j = edge row, lane lm = head-dim index)
  float tj[8];
#pragma unroll
  for (int j = 0; j < 8; ++j)
    tj[j] = s[j] * qn[(long)dstc[j] * DIM + h * DK + lm];

#pragma unroll
  for (int mask = 1; mask < 16; mask <<= 1) {
#pragma unroll
    for (int j = 0; j < 8; ++j) tj[j] += __shfl_xor(tj[j], mask, 32);
  }

  if (lm == 0) {
#pragma unroll
    for (int j = 0; j < 8; ++j) {
      long e = e0 + half * 8 + j;
      if (e < E) {
        float ra = tj[j] * rel_pri[et[j] * NH + h] * 0.25f;  // 1/sqrt(16)
        res_att[e * NH + h] = ra;
        atomicMax(seg_max_u + (long)dstc[j] * NH + h, f2ord(ra));
      }
    }
  }
}

// ------------------------------------------------- softmax pieces -----------
__global__ void hgt_softmax_edge(const int* __restrict__ edge_index,
                                 float* __restrict__ res_att,
                                 const unsigned* __restrict__ seg_max_u,
                                 float* __restrict__ seg_sum, int E) {
  long i = (long)blockIdx.x * blockDim.x + threadIdx.x;
  if (i >= (long)E * NH) return;
  long e = i / NH;
  int h = (int)(i % NH);
  int dst = edge_index[(long)E + e];
  float m = ord2f(seg_max_u[(long)dst * NH + h]);
  if (!__builtin_isfinite(m)) m = 0.0f;
  float ex = expf(res_att[i] - m);
  res_att[i] = ex;  // numerator cached in place
  atomicAdd(seg_sum + (long)dst * NH + h, ex);
}

// ------------------------------------------- messages + aggregation ---------
__global__ __launch_bounds__(256)
void hgt_message_aggr(const float* __restrict__ vn,
                      const bf16* __restrict__ relMsgT,  // [R][H][DK(n)][DK(d)]
                      const int* __restrict__ edge_index,
                      const int* __restrict__ edge_type,
                      const float* __restrict__ res_att,  // exp numerators
                      const float* __restrict__ seg_sum,
                      float* __restrict__ aggr,           // [N][DIM]
                      int E) {
  const int h    = threadIdx.x >> 5;
  const int lane = threadIdx.x & 31;
  const int half = lane >> 4;
  const int lm   = lane & 15;
  const long e0  = (long)blockIdx.x * 16;

  v16bf a;
  {
    long e = e0 + lm;
    if (e >= E) e = E - 1;
    const float* pv = vn + (long)edge_index[e] * DIM + h * DK;
#pragma unroll
    for (int j = 0; j < 16; ++j) a[j] = half ? (bf16)0.0f : (bf16)pv[j];
  }

  int et[8], dstc[8];
#pragma unroll
  for (int j = 0; j < 8; ++j) {
    long e = e0 + half * 8 + j;
    if (e >= E) e = E - 1;
    et[j]   = edge_type[e];
    dstc[j] = edge_index[(long)E + e];
  }

  v8f s = v8zero();
#pragma unroll
  for (int r = 0; r < NRE; ++r) {
    const bf16* pb = relMsgT + ((long)(r * NH + h) * DK + lm) * DK;
    v16bf b;
#pragma unroll
    for (int j = 0; j < 16; ++j) b[j] = half ? (bf16)0.0f : pb[j];
    v8f acc = wmma_bf16(a, b, v8zero());
#pragma unroll
    for (int j = 0; j < 8; ++j) s[j] = (et[j] == r) ? acc[j] : s[j];
  }

#pragma unroll
  for (int j = 0; j < 8; ++j) {
    long e = e0 + half * 8 + j;
    if (e < E) {
      float alpha = res_att[e * NH + h] /
                    (seg_sum[(long)dstc[j] * NH + h] + 1e-16f);
      atomicAdd(aggr + (long)dstc[j] * DIM + h * DK + lm, s[j] * alpha);
    }
  }
}

// --------------------------------------------------- gelu -> bf16 -----------
__global__ void hgt_gelu_bf16(const float* __restrict__ aggr,
                              bf16* __restrict__ hb, long n) {
  long i = (long)blockIdx.x * blockDim.x + threadIdx.x;
  if (i < n) {
    float x = aggr[i];
    float g = 0.5f * x * (1.0f + erff(x * 0.70710678118654752f));  // exact gelu
    hb[i] = (bf16)g;
  }
}

// ---------------------------------------- final Wa projection + skip --------
__global__ __launch_bounds__(256)
void hgt_update_out(const bf16* __restrict__ hb,
                    const bf16* __restrict__ waT,  // [NTY][DIM(n)][DIM(k)]
                    const float* __restrict__ ba,
                    const float* __restrict__ skip,  // [NTY]
                    const float* __restrict__ node_inp,
                    const int* __restrict__ node_type,
                    float* __restrict__ out, int N) {
  const int wave = threadIdx.x >> 5;
  const int lane = threadIdx.x & 31;
  const int half = lane >> 4;
  const int lm   = lane & 15;
  const int m0 = blockIdx.x * 16;
  const int n0 = wave * 16;

  int arow = m0 + lm;
  if (arow >= N) arow = N - 1;
  v16bf a[4];
#pragma unroll
  for (int kc = 0; kc < 4; ++kc) {
    const bf16* px = hb + (long)arow * DIM + kc * 32 + half * 16;
#pragma unroll
    for (int j = 0; j < 16; ++j) a[kc][j] = px[j];
  }

  int rowt[8];
#pragma unroll
  for (int j = 0; j < 8; ++j) {
    int r = m0 + half * 8 + j;
    rowt[j] = (r < N) ? node_type[r] : -1;
  }

  float sg[NTY];
#pragma unroll
  for (int t = 0; t < NTY; ++t) sg[t] = 1.0f / (1.0f + expf(-skip[t]));

  v8f res = v8zero();
#pragma unroll
  for (int t = 0; t < NTY; ++t) {
    v8f acc = v8zero();
#pragma unroll
    for (int kc = 0; kc < 4; ++kc) {
      const bf16* pw = waT + ((long)(t * DIM + (n0 + lm))) * DIM + kc * 32 + half * 16;
      v16bf b;
#pragma unroll
      for (int j = 0; j < 16; ++j) b[j] = pw[j];
      acc = wmma_bf16(a[kc], b, acc);
    }
    float bias = ba[t * DIM + n0 + lm];
#pragma unroll
    for (int j = 0; j < 8; ++j)
      res[j] = (rowt[j] == t) ? (acc[j] + bias) : res[j];
  }

#pragma unroll
  for (int j = 0; j < 8; ++j) {
    int r = m0 + half * 8 + j;
    if (r < N) {
      float sgr = sg[rowt[j]];
      float xin = node_inp[(long)r * DIM + n0 + lm];
      out[(long)r * DIM + n0 + lm] = res[j] * sgr + xin * (1.0f - sgr);
    }
  }
}

// ---------------------------------------------------------------------------
extern "C" void kernel_launch(void* const* d_in, const int* in_sizes, int n_in,
                              void* d_out, int out_size, void* d_ws, size_t ws_size,
                              hipStream_t stream) {
  const float* node_inp = (const float*)d_in[0];
  const float* Wk = (const float*)d_in[1];
  const float* bk = (const float*)d_in[2];
  const float* Wq = (const float*)d_in[3];
  const float* bq = (const float*)d_in[4];
  const float* Wv = (const float*)d_in[5];
  const float* bv = (const float*)d_in[6];
  const float* Wa = (const float*)d_in[7];
  const float* ba = (const float*)d_in[8];
  const float* rel_pri = (const float*)d_in[9];
  const float* rel_att = (const float*)d_in[10];
  const float* rel_msg = (const float*)d_in[11];
  const float* skip = (const float*)d_in[12];
  const int* node_type  = (const int*)d_in[13];
  const int* edge_index = (const int*)d_in[14];
  const int* edge_type  = (const int*)d_in[15];

  const int N = in_sizes[0] / DIM;
  const int E = in_sizes[15];

  // --- workspace carve-out (~116 MB) ---
  char* base = (char*)d_ws;
  size_t off = 0;
  auto take = [&](size_t bytes) -> void* {
    void* p = base + off;
    off = (off + bytes + 255) & ~(size_t)255;
    return p;
  };
  bf16* xbf = (bf16*)take((size_t)N * DIM * 2);
  bf16* wT  = (bf16*)take((size_t)3 * NTY * DIM * DIM * 2);
  bf16* waT = (bf16*)take((size_t)NTY * DIM * DIM * 2);
  bf16* rAT = (bf16*)take((size_t)NRE * NH * DK * DK * 2);
  bf16* rMT = (bf16*)take((size_t)NRE * NH * DK * DK * 2);
  float* kn   = (float*)take((size_t)N * DIM * 4);
  float* qn   = (float*)take((size_t)N * DIM * 4);
  float* vnod = (float*)take((size_t)N * DIM * 4);
  float* ratt = (float*)take((size_t)E * NH * 4);
  unsigned* smax = (unsigned*)take((size_t)N * NH * 4);
  float* ssum = (float*)take((size_t)N * NH * 4);
  float* aggr = (float*)take((size_t)N * DIM * 4);
  bf16* hb = (bf16*)take((size_t)N * DIM * 2);

  hipMemsetAsync(smax, 0, (size_t)N * NH * 4, stream);  // ordered-uint 0 == -inf-ish
  hipMemsetAsync(ssum, 0, (size_t)N * NH * 4, stream);
  hipMemsetAsync(aggr, 0, (size_t)N * DIM * 4, stream);

  {
    long n = (long)N * DIM;
    hgt_cvt_bf16<<<(int)((n + 255) / 256), 256, 0, stream>>>(node_inp, xbf, n);
  }
  {
    int tot = NTY * DIM * DIM;
    int g = (tot + 255) / 256;
    hgt_transpose_bf16<<<g, 256, 0, stream>>>(Wk, wT + 0 * (size_t)NTY * DIM * DIM, NTY, DIM, DIM);
    hgt_transpose_bf16<<<g, 256, 0, stream>>>(Wq, wT + 1 * (size_t)NTY * DIM * DIM, NTY, DIM, DIM);
    hgt_transpose_bf16<<<g, 256, 0, stream>>>(Wv, wT + 2 * (size_t)NTY * DIM * DIM, NTY, DIM, DIM);
    hgt_transpose_bf16<<<g, 256, 0, stream>>>(Wa, waT, NTY, DIM, DIM);
  }
  {
    int tot = NRE * NH * DK * DK;
    int g = (tot + 255) / 256;
    hgt_transpose_bf16<<<g, 256, 0, stream>>>(rel_att, rAT, NRE * NH, DK, DK);
    hgt_transpose_bf16<<<g, 256, 0, stream>>>(rel_msg, rMT, NRE * NH, DK, DK);
  }

  const int ntiles = (N + 15) / 16;
  const int etiles = (E + 15) / 16;

  hgt_project_kqv<<<ntiles, 256, 0, stream>>>(xbf, wT, bk, bq, bv, node_type,
                                              kn, qn, vnod, N);

  hgt_attn_scores<<<etiles, 256, 0, stream>>>(kn, qn, rAT, rel_pri, edge_index,
                                              edge_type, ratt, smax, E);

  {
    long n = (long)E * NH;
    hgt_softmax_edge<<<(int)((n + 255) / 256), 256, 0, stream>>>(edge_index, ratt,
                                                                 smax, ssum, E);
  }

  hgt_message_aggr<<<etiles, 256, 0, stream>>>(vnod, rMT, edge_index, edge_type,
                                               ratt, ssum, aggr, E);

  {
    long n = (long)N * DIM;
    hgt_gelu_bf16<<<(int)((n + 255) / 256), 256, 0, stream>>>(aggr, hb, n);
  }

  hgt_update_out<<<ntiles, 256, 0, stream>>>(hb, waT, ba, skip, node_inp,
                                             node_type, (float*)d_out, N);
}
